// L2neighs_Aggregator_20375324852399
// MI455X (gfx1250) — compile-verified, matrix operand order checked by reading, not run on hardware
//
#include <hip/hip_runtime.h>
#include <hip/hip_bf16.h>

// ---------------------------------------------------------------------------
// Fused L2-neighbor attention aggregator for MI455X (gfx1250, wave32, WMMA).
// One block (256 threads = 8 wave32) handles one node: gather -> bf16 X in
// LDS -> 4 chained V_WMMA_F32_16X16X32_BF16 GEMMs -> on-chip softmax ->
// attention-weighted reduce. Weights are pre-converted to bf16 + transposed
// (N-major) in d_ws so every WMMA B-fragment is two contiguous 16B loads.
// ---------------------------------------------------------------------------

typedef __attribute__((ext_vector_type(16))) __bf16 v16bf;
typedef __attribute__((ext_vector_type(8)))  float  v8f;

#define NNODES 4096
#define KPATHS 64
#define NATTR  8
#define DIM    128

// LDS row strides (bf16 elements), padded to stay 16B-aligned & spread banks
#define SX 520   // X  : 64 x 512 (+8)
#define S1 264   // H1 : 64 x 256 (+8)
#define SH 136   // H  : 64 x 128 (+8)

__device__ __forceinline__ v8f zero_v8f() {
  v8f z = {0.f, 0.f, 0.f, 0.f, 0.f, 0.f, 0.f, 0.f};
  return z;
}

// 16-bit A/B fragment per CDNA5 ISA layout (16x32 / 32x16 bf16):
// lane half h, idx = lane%16 holds elems 0..7 = M[idx][k0+8h .. +7]
//                          and elems 8..15 = M[idx][k0+16+8h .. +7]
// caller passes p = &M[idx][k0 + 8h]; second chunk is +16 elements.
__device__ __forceinline__ v16bf load_frag(const __bf16* p) {
  union { v16bf v; uint4 q[2]; } f;
  f.q[0] = *(const uint4*)(p);
  f.q[1] = *(const uint4*)(p + 16);
  return f.v;
}

// One wave computes NT 16x16 output tiles (same mtile) over KSTEPS k-steps.
// A: LDS bf16, row-major [M][lda]. B: bf16 N-major (transposed weights) [N][ldb].
template<int KSTEPS, int NT>
__device__ __forceinline__ void gemm_tiles(const __bf16* A, int lda,
                                           const __bf16* B, int ldb,
                                           int mtile, int ntile0, int lane,
                                           v8f* acc) {
  const int half = lane >> 4;
  const int l16  = lane & 15;
  const int arow = mtile * 16 + l16;
  for (int ks = 0; ks < KSTEPS; ++ks) {
    const int koff = ks * 32 + half * 8;
    v16bf af = load_frag(A + arow * lda + koff);
#pragma unroll
    for (int t = 0; t < NT; ++t) {
      const int ncol = (ntile0 + t) * 16 + l16;
      v16bf bfr = load_frag(B + ncol * ldb + koff);
      acc[t] = __builtin_amdgcn_wmma_f32_16x16x32_bf16(
          false, af, false, bfr, (short)0, acc[t], false, false);
    }
  }
}

// D-matrix layout: VGPR v, lane half h, col = l16 -> element [v + 8h][col]
__device__ __forceinline__ void store_tile_relu_bf16(__bf16* dst, int ldd,
                                                     int mtile, int ncol, int half,
                                                     const v8f& acc, float bias) {
#pragma unroll
  for (int v = 0; v < 8; ++v) {
    const int row = mtile * 16 + half * 8 + v;
    dst[row * ldd + ncol] = (__bf16)fmaxf(acc[v] + bias, 0.f);
  }
}

__device__ __forceinline__ void store_bf16x4(__bf16* dst, float4 v) {
  union { __bf16 b[4]; uint2 u; } t;
  t.b[0] = (__bf16)v.x; t.b[1] = (__bf16)v.y;
  t.b[2] = (__bf16)v.z; t.b[3] = (__bf16)v.w;
  *(uint2*)dst = t.u;
}

// ---------------------------------------------------------------------------
// Weight prep: f32 row-major [R][C]  ->  bf16 transposed [C][R] in workspace.
// ---------------------------------------------------------------------------
__global__ __launch_bounds__(256) void convert_transpose_bf16(
    const float* __restrict__ src, __bf16* __restrict__ dst, int R, int C) {
  const int i = blockIdx.x * 256 + threadIdx.x;
  if (i >= R * C) return;
  const int r = i / C, c = i % C;
  dst[c * R + r] = (__bf16)src[i];
}

// ---------------------------------------------------------------------------
// Main fused kernel: one node per block.
// ---------------------------------------------------------------------------
__global__ __launch_bounds__(256) void hinsage_l2_agg(
    const int* __restrict__ nodes,
    const int* __restrict__ paths_rel,
    const int* __restrict__ paths_nbr,
    const int* __restrict__ attrs,
    const float* __restrict__ u2e,
    const float* __restrict__ r2e,
    const float* __restrict__ ua2e,
    const float* __restrict__ b1,
    const float* __restrict__ b2,
    const float* __restrict__ ab1,
    const float* __restrict__ ab2,
    const float* __restrict__ A3,
    const float* __restrict__ ab3,
    const __bf16* __restrict__ w1t,   // [256][512]
    const __bf16* __restrict__ w2t,   // [128][256]
    const __bf16* __restrict__ a1t,   // [128][256]
    const __bf16* __restrict__ a2t,   // [128][128]
    float* __restrict__ out)
{
  extern __shared__ char smem[];
  __bf16* xs  = (__bf16*)smem;            // [64][SX]
  __bf16* h1s = xs  + KPATHS * SX;        // [64][S1]
  __bf16* hs  = h1s + KPATHS * S1;        // [64][SH]
  __bf16* a1s = h1s;                      // alias: H1 dead after GEMM2
  float* selfe  = (float*)(hs + KPATHS * SH);   // [128]
  float* bias1p = selfe + DIM;                  // [128]
  float* logits = bias1p + DIM;                 // [64]
  float* wts    = logits + KPATHS;              // [64]

  const int node = blockIdx.x;
  const int tid  = threadIdx.x;
  const int lane = tid & 31;
  const int wave = tid >> 5;

  // ---- Phase 0: gather embeddings -> X bf16; self embedding; logits init
  {
    const int row = tid >> 2;      // 64 rows
    const int seg = tid & 3;       // 4 x 128-col segments
    __bf16* xrow = xs + row * SX + seg * DIM;
    if (seg < 2) {                 // r1 / r2
      const int rid = paths_rel[(node * KPATHS + row) * 2 + seg];
      const float4* src = (const float4*)(r2e + (size_t)rid * DIM);
#pragma unroll 4
      for (int c = 0; c < DIM / 4; ++c) store_bf16x4(xrow + c * 4, src[c]);
    } else if (seg == 2) {         // neighbor embedding
      const int nid = paths_nbr[node * KPATHS + row];
      const float4* src = (const float4*)(u2e + (size_t)nid * DIM);
#pragma unroll 4
      for (int c = 0; c < DIM / 4; ++c) store_bf16x4(xrow + c * 4, src[c]);
    } else {                       // sum of 8 attribute embeddings
      int idx[NATTR];
#pragma unroll
      for (int a = 0; a < NATTR; ++a)
        idx[a] = attrs[(node * KPATHS + row) * NATTR + a];
      for (int c = 0; c < DIM / 4; ++c) {
        float4 s = make_float4(0.f, 0.f, 0.f, 0.f);
#pragma unroll
        for (int a = 0; a < NATTR; ++a) {
          float4 e = *(const float4*)(ua2e + (size_t)idx[a] * DIM + c * 4);
          s.x += e.x; s.y += e.y; s.z += e.z; s.w += e.w;
        }
        store_bf16x4(xrow + c * 4, s);
      }
    }
    if (tid < DIM)    selfe[tid]  = u2e[(size_t)nodes[node] * DIM + tid];
    if (tid < KPATHS) logits[tid] = ab3[0];
  }
  __syncthreads();

  // ---- Fold broadcast self-half of attention concat into a per-node bias:
  //      bias1p[d] = ab1[d] + sum_j selfe[j] * A1[128+j][d]
  if (tid < DIM) {
    float s = ab1[tid];
    const __bf16* arow = a1t + tid * 256 + DIM;   // k = 128..255 of A1^T row
#pragma unroll 8
    for (int j = 0; j < DIM; ++j) s += selfe[j] * (float)arow[j];
    bias1p[tid] = s;
  }

  const int half = lane >> 4;
  const int l16  = lane & 15;

  // ---- GEMM1: H1 = relu(X[64,512] @ W1[512,256] + b1)   (4 mt x 16 nt)
  {
    const int mtile  = wave >> 1;
    const int ntile0 = (wave & 1) * 8;
    v8f acc[8];
#pragma unroll
    for (int t = 0; t < 8; ++t) acc[t] = zero_v8f();
    gemm_tiles<16, 8>(xs, SX, w1t, 512, mtile, ntile0, lane, acc);
#pragma unroll
    for (int t = 0; t < 8; ++t) {
      const int ncol = (ntile0 + t) * 16 + l16;
      store_tile_relu_bf16(h1s, S1, mtile, ncol, half, acc[t], b1[ncol]);
    }
  }
  __syncthreads();

  // ---- GEMM2: H = relu(H1[64,256] @ W2[256,128] + b2)   (4 mt x 8 nt)
  {
    const int mtile  = wave >> 1;
    const int ntile0 = (wave & 1) * 4;
    v8f acc[4];
#pragma unroll
    for (int t = 0; t < 4; ++t) acc[t] = zero_v8f();
    gemm_tiles<8, 4>(h1s, S1, w2t, 256, mtile, ntile0, lane, acc);
#pragma unroll
    for (int t = 0; t < 4; ++t) {
      const int ncol = (ntile0 + t) * 16 + l16;
      store_tile_relu_bf16(hs, SH, mtile, ncol, half, acc[t], b2[ncol]);
    }
  }
  __syncthreads();

  // ---- GEMM3: a1 = relu(H[64,128] @ A1[0:128,:] + bias1p)
  {
    const int mtile  = wave >> 1;
    const int ntile0 = (wave & 1) * 4;
    v8f acc[4];
#pragma unroll
    for (int t = 0; t < 4; ++t) acc[t] = zero_v8f();
    gemm_tiles<4, 4>(hs, SH, a1t, 256, mtile, ntile0, lane, acc);
#pragma unroll
    for (int t = 0; t < 4; ++t) {
      const int ncol = (ntile0 + t) * 16 + l16;
      store_tile_relu_bf16(a1s, SH, mtile, ncol, half, acc[t], bias1p[ncol]);
    }
  }
  __syncthreads();

  // ---- GEMM4: logits += (relu(a1 @ A2 + ab2)) . A3  (a2 never materialized)
  {
    const int mtile  = wave >> 1;
    const int ntile0 = (wave & 1) * 4;
    v8f acc[4];
#pragma unroll
    for (int t = 0; t < 4; ++t) acc[t] = zero_v8f();
    gemm_tiles<4, 4>(a1s, SH, a2t, 128, mtile, ntile0, lane, acc);
#pragma unroll
    for (int t = 0; t < 4; ++t) {
      const int ncol = (ntile0 + t) * 16 + l16;
      const float ab2c = ab2[ncol];
      const float a3c  = A3[ncol];
#pragma unroll
      for (int v = 0; v < 8; ++v) {
        const int row = mtile * 16 + half * 8 + v;
        const float val = fmaxf(acc[t][v] + ab2c, 0.f);
        atomicAdd(&logits[row], val * a3c);      // ds_add_f32
      }
    }
  }
  __syncthreads();

  // ---- Softmax over K=64 logits (single wave32, shuffle reductions)
  if (tid < 32) {
    float l0 = logits[tid], l1 = logits[tid + 32];
    float m = fmaxf(l0, l1);
#pragma unroll
    for (int o = 16; o > 0; o >>= 1) m = fmaxf(m, __shfl_xor(m, o, 32));
    float e0 = __expf(l0 - m), e1 = __expf(l1 - m);
    float s = e0 + e1;
#pragma unroll
    for (int o = 16; o > 0; o >>= 1) s += __shfl_xor(s, o, 32);
    const float inv = 1.f / s;
    wts[tid] = e0 * inv;
    wts[tid + 32] = e1 * inv;
  }
  __syncthreads();

  // ---- out[n,d] = sum_k wts[k] * H[k,d]
  if (tid < DIM) {
    float accv = 0.f;
#pragma unroll 8
    for (int k = 0; k < KPATHS; ++k)
      accv += wts[k] * (float)hs[k * SH + tid];
    out[(size_t)node * DIM + tid] = accv;
  }
}

// ---------------------------------------------------------------------------
extern "C" void kernel_launch(void* const* d_in, const int* in_sizes, int n_in,
                              void* d_out, int out_size, void* d_ws, size_t ws_size,
                              hipStream_t stream) {
  const int*   nodes     = (const int*)  d_in[0];
  const int*   paths_rel = (const int*)  d_in[1];
  const int*   paths_nbr = (const int*)  d_in[2];
  const int*   attrs     = (const int*)  d_in[3];
  const float* u2e       = (const float*)d_in[4];
  const float* r2e       = (const float*)d_in[5];
  const float* ua2e      = (const float*)d_in[6];
  const float* W1        = (const float*)d_in[7];   // [512][256]
  const float* b1        = (const float*)d_in[8];
  const float* W2        = (const float*)d_in[9];   // [256][128]
  const float* b2        = (const float*)d_in[10];
  const float* A1        = (const float*)d_in[11];  // [256][128]
  const float* ab1       = (const float*)d_in[12];
  const float* A2        = (const float*)d_in[13];  // [128][128]
  const float* ab2       = (const float*)d_in[14];
  const float* A3        = (const float*)d_in[15];  // [128][1]
  const float* ab3       = (const float*)d_in[16];
  float* out = (float*)d_out;

  // Workspace layout (bf16, transposed weights, ~416 KB total)
  __bf16* w1t = (__bf16*)d_ws;          // 256 x 512
  __bf16* w2t = w1t + 256 * 512;        // 128 x 256
  __bf16* a1t = w2t + 128 * 256;        // 128 x 256
  __bf16* a2t = a1t + 128 * 256;        // 128 x 128

  convert_transpose_bf16<<<(512 * 256 + 255) / 256, 256, 0, stream>>>(W1, w1t, 512, 256);
  convert_transpose_bf16<<<(256 * 128 + 255) / 256, 256, 0, stream>>>(W2, w2t, 256, 128);
  convert_transpose_bf16<<<(256 * 128 + 255) / 256, 256, 0, stream>>>(A1, a1t, 256, 128);
  convert_transpose_bf16<<<(128 * 128 + 255) / 256, 256, 0, stream>>>(A2, a2t, 128, 128);

  constexpr size_t lds_bytes =
      (size_t)(KPATHS * SX + KPATHS * S1 + KPATHS * SH) * 2 +   // bf16 tiles
      (size_t)(DIM + DIM + KPATHS + KPATHS) * 4;                // f32 scratch
  // = 119,296 B -> 2 workgroups per 320 KB WGP

  hinsage_l2_agg<<<NNODES, 256, lds_bytes, stream>>>(
      nodes, paths_rel, paths_nbr, attrs, u2e, r2e, ua2e,
      b1, b2, ab1, ab2, A3, ab3, w1t, w2t, a1t, a2t, out);
}
